// AudioQuantizer_70153995812935
// MI455X (gfx1250) — matrix operand order repaired and checked by reading, not compiled, not run on hardware
//
#include <hip/hip_runtime.h>

// ---- problem constants (from reference) ----
#define BB   8
#define TT   2048
#define DD   1024
#define QQ   8
#define KK   1024
#define DSS  128

typedef __attribute__((ext_vector_type(16))) __bf16 v16bf;
typedef __attribute__((ext_vector_type(8)))  float  v8f;

static constexpr int  NTOK  = BB * TT;        // 16384 tokens
static constexpr int  NTILE = NTOK / 16;      // 1024 tiles of 16 tokens
static constexpr int  NGRP  = NTILE / 2;      // 512 groups (2 tiles / wave pass)
static constexpr int  GRID  = 256;

// output layout: quant_t | idx_t | quant_d | idx_d (flat, return order)
static constexpr long OFF_QT = 0;
static constexpr long OFF_IT = (long)NTOK * DD;               // 16777216
static constexpr long OFF_QD = OFF_IT + (long)NTOK * QQ;      // 16908288
static constexpr long OFF_ID = OFF_QD + (long)NTOK * DD;      // 33685504

// native f32 -> bf16 (lowers to v_cvt_bf16 / v_cvt_pk_bf16_f32 on gfx1250)
__device__ __forceinline__ __bf16 f2bf(float f) { return (__bf16)f; }

__global__ __launch_bounds__(256)
void audio_vq_kernel(const float* __restrict__ x,
                     const float* __restrict__ cb_t,
                     const float* __restrict__ cb_d,
                     float* __restrict__ out) {
  // c2[cb][q][k] = sum_d cb[q][k][d]^2  (64 KB LDS)
  __shared__ float c2s[2 * QQ * KK];
  const int tid = threadIdx.x;
  for (int e = tid; e < 2 * QQ * KK; e += 256) {
    const float* src = (e < QQ * KK) ? (cb_t + (long)e * DSS)
                                     : (cb_d + (long)(e - QQ * KK) * DSS);
    float s = 0.f;
#pragma unroll 4
    for (int j = 0; j < DSS; ++j) { float v = src[j]; s += v * v; }
    c2s[e] = s;
  }
  __syncthreads();

  const int wid  = tid >> 5;       // wave id == q (8 waves)
  const int lane = tid & 31;
  const int hl   = lane >> 4;      // half: 0 = lanes 0-15, 1 = lanes 16-31
  const int ln   = lane & 15;
  const int q    = wid;

  for (int g = blockIdx.x; g < NGRP; g += gridDim.x) {
    const int tile0 = g * 2;

    // ---------- load A = (-2*x) as bf16 for 2 tiles, plus per-row x^2 ----------
    // A 16-bit 16x32 layout (ISA 7.12.2): lane row M = ln;
    //   vgpr0-3: K = hl*8 + [0..7]; vgpr4-7: K = 16 + hl*8 + [0..7]
    v16bf A[2][4];
    float x2v[2][8];
    float myx2 = 0.f;
    {
      // lane L computes x2 of global row tile0*16 + L (32 rows total)
      const float* xr = x + ((long)(tile0 * 16 + lane)) * DD + q * DSS;
#pragma unroll 4
      for (int j = 0; j < DSS; ++j) { float v = xr[j]; myx2 += v * v; }
    }
#pragma unroll
    for (int t = 0; t < 2; ++t) {
      const float* xr = x + ((long)(tile0 * 16 + t * 16 + ln)) * DD + q * DSS;
#pragma unroll
      for (int kk = 0; kk < 4; ++kk) {
        const int b0 = kk * 32 + hl * 8;
        const int b1 = kk * 32 + 16 + hl * 8;
        v16bf a;
#pragma unroll
        for (int i = 0; i < 8; ++i) a[i]     = f2bf(-2.f * xr[b0 + i]);
#pragma unroll
        for (int i = 0; i < 8; ++i) a[8 + i] = f2bf(-2.f * xr[b1 + i]);
        A[t][kk] = a;
      }
      // C/D layout: this lane's slot r holds row M = r + 8*hl of tile t
#pragma unroll
      for (int r = 0; r < 8; ++r)
        x2v[t][r] = __shfl(myx2, t * 16 + hl * 8 + r, 32);
    }

    // ---------- both codebooks ----------
#pragma unroll 1
    for (int c = 0; c < 2; ++c) {
      const float* cbp = c ? cb_d : cb_t;
      const float* c2p = c2s + c * QQ * KK + q * KK;

      float minv[2][8];
      int   mini[2][8];
#pragma unroll
      for (int t = 0; t < 2; ++t)
#pragma unroll
        for (int i = 0; i < 8; ++i) { minv[t][i] = 3.4e38f; mini[t][i] = 0; }

#pragma unroll 1
      for (int nt = 0; nt < KK / 16; ++nt) {
        // B 16-bit 32x16 layout: lane column N = ln; K = hl*8 + [0..7],
        // then K = 16 + hl*8 + [0..7] (same order as A).
        const float* br = cbp + ((long)(q * KK + nt * 16 + ln)) * DSS + hl * 8;
        if (nt + 1 < KK / 16)
          __builtin_prefetch(br + 16 * DSS, 0, 1);  // next N-tile -> global_prefetch_b8
        v16bf Bv[4];
#pragma unroll
        for (int kk = 0; kk < 4; ++kk) {
          v16bf b;
#pragma unroll
          for (int i = 0; i < 16; ++i) b[i] = f2bf(br[kk * 32 + (i & 7) + (i >> 3) * 16]);
          Bv[kk] = b;
        }
        const float c2n  = c2p[nt * 16 + ln];  // per-column c^2
        const int   nidx = nt * 16 + ln;
#pragma unroll
        for (int t = 0; t < 2; ++t) {
          v8f acc;
#pragma unroll
          for (int i = 0; i < 8; ++i) acc[i] = c2n;  // C := c^2 (per column)
#pragma unroll
          for (int kk = 0; kk < 4; ++kk)
            acc = __builtin_amdgcn_wmma_f32_16x16x32_bf16(
                false, A[t][kk], false, Bv[kk], (short)0, acc, false, false);
          // acc = c^2 - 2*x.c ; d2 = acc + x^2 (monotone in true distance)
#pragma unroll
          for (int i = 0; i < 8; ++i) {
            float d2 = acc[i] + x2v[t][i];
            if (d2 < minv[t][i]) { minv[t][i] = d2; mini[t][i] = nidx; }
          }
        }
      }

      // cross-lane argmin over the 16 columns (xor stays inside each half)
#pragma unroll
      for (int t = 0; t < 2; ++t)
#pragma unroll
        for (int i = 0; i < 8; ++i) {
          float v  = minv[t][i];
          int   id = mini[t][i];
#pragma unroll
          for (int m = 8; m >= 1; m >>= 1) {
            float ov = __shfl_xor(v, m, 32);
            int   oi = __shfl_xor(id, m, 32);
            if (ov < v || (ov == v && oi < id)) { v = ov; id = oi; }
          }
          minv[t][i] = v; mini[t][i] = id;
        }

      float* qout = out + (c ? OFF_QD : OFF_QT);
      float* iout = out + (c ? OFF_ID : OFF_IT);
#pragma unroll
      for (int t = 0; t < 2; ++t) {
        const long tokbase = (long)(tile0 + t) * 16;
        // indices: lanes 0-15 hold rows 0-7 (slot r), lanes 16-31 rows 8-15
#pragma unroll
        for (int r = 0; r < 8; ++r) {
          if (lane == 0)  iout[(tokbase + r)     * QQ + q] = (float)mini[t][r];
          if (lane == 16) iout[(tokbase + 8 + r) * QQ + q] = (float)mini[t][r];
        }
        // gather the chosen fp32 codebook rows: 128 floats = 32 lanes x float4
#pragma unroll
        for (int m = 0; m < 16; ++m) {
          int id = __shfl(mini[t][m & 7], (m < 8) ? 0 : 16, 32);
          const float4* srcr = (const float4*)(cbp + ((long)(q * KK + id)) * DSS);
          float4*       dstr = (float4*)(qout + (tokbase + m) * DD + q * DSS);
          dstr[lane] = srcr[lane];
        }
      }
    }
  }
}

extern "C" void kernel_launch(void* const* d_in, const int* in_sizes, int n_in,
                              void* d_out, int out_size, void* d_ws, size_t ws_size,
                              hipStream_t stream) {
  (void)in_sizes; (void)n_in; (void)d_ws; (void)ws_size; (void)out_size;
  const float* x    = (const float*)d_in[0];
  const float* cb_t = (const float*)d_in[1];
  const float* cb_d = (const float*)d_in[2];
  float* out = (float*)d_out;
  audio_vq_kernel<<<GRID, 256, 0, stream>>>(x, cb_t, cb_d, out);
}
